// MambaBlock_68281390071812
// MI455X (gfx1250) — matrix operand order repaired and checked by reading, not compiled, not run on hardware
//
#include <hip/hip_runtime.h>
#include <hip/hip_bf16.h>

// ---------------- problem constants (match reference) ----------------
#define D_MODEL   256
#define D_STATE   16
#define D_CONV    4
#define D_INNER   512          // EXPAND * D_MODEL
#define DT_RANK   16
#define B_SZ      8
#define L_SEQ     2048
#define N_LAYERS  3
#define M_TOTAL   (B_SZ * L_SEQ)   // 16384 rows for all GEMMs

typedef __attribute__((ext_vector_type(16))) _Float16 v16h;
typedef __attribute__((ext_vector_type(8)))  float    v8f;

// gfx1250 async global->LDS path (ASYNCcnt), if this toolchain exposes it
#if defined(__has_builtin)
#if __has_builtin(__builtin_amdgcn_global_load_async_to_lds_b128)
#define HAVE_ASYNC_LDS 1
#endif
#endif
#ifndef HAVE_ASYNC_LDS
#define HAVE_ASYNC_LDS 0
#endif

#if HAVE_ASYNC_LDS
// signature (probe-verified): param0 = int __vector_size__(16)* in AS1 (device),
// param1 = AS3 pointer, + imm offset + imm cpol (4 args).
typedef int v4i __attribute__((vector_size(16)));
typedef __attribute__((address_space(1))) v4i* gptr_b128;
typedef __attribute__((address_space(3))) v4i* lptr_b128;
#endif

// ======================================================================
// fp32 -> fp16 weight conversion (weights are tiny; done right before use)
// ======================================================================
__global__ void k_f32_to_f16(const float* __restrict__ src,
                             _Float16* __restrict__ dst, int n) {
  int i = blockIdx.x * blockDim.x + threadIdx.x;
  if (i < n) dst[i] = (_Float16)src[i];
}

// ======================================================================
// LayerNorm over D_MODEL=256; one row per 256-thread block; f16 output
// ======================================================================
__global__ __launch_bounds__(256)
void k_layernorm(const float* __restrict__ x, const float* __restrict__ gw,
                 const float* __restrict__ bw, _Float16* __restrict__ out) {
  __shared__ float sred[8];
  __shared__ float sstat[2];
  const int t = threadIdx.x;
  const size_t row = blockIdx.x;
  float v = x[row * D_MODEL + t];

  float s = v;
  #pragma unroll
  for (int off = 16; off >= 1; off >>= 1) s += __shfl_xor(s, off, 32);
  if ((t & 31) == 0) sred[t >> 5] = s;
  __syncthreads();
  if (t == 0) {
    float tt = 0.f;
    #pragma unroll
    for (int i = 0; i < 8; ++i) tt += sred[i];
    sstat[0] = tt * (1.f / D_MODEL);
  }
  __syncthreads();
  const float mu = sstat[0];
  const float d  = v - mu;
  s = d * d;
  #pragma unroll
  for (int off = 16; off >= 1; off >>= 1) s += __shfl_xor(s, off, 32);
  if ((t & 31) == 0) sred[t >> 5] = s;
  __syncthreads();
  if (t == 0) {
    float tt = 0.f;
    #pragma unroll
    for (int i = 0; i < 8; ++i) tt += sred[i];
    sstat[1] = tt * (1.f / D_MODEL);
  }
  __syncthreads();
  const float r = rsqrtf(sstat[1] + 1e-5f);
  out[row * D_MODEL + t] = (_Float16)(d * r * gw[t] + bw[t]);
}

// ======================================================================
// WMMA GEMM:  C[M,N](f32) = A[M,K](f16) * W[N,K](f16)^T  (+ addend)
//   block = 256 threads (8 waves); block tile 128M x 64N, K-tile 64
//   wave grid 4(M) x 2(N); each wave: 32x32 register tile = 4 v8f accs
//   Double-buffered LDS + GLOBAL_LOAD_ASYNC_TO_LDS_B128 pipeline:
//   tile k+1 DMA (ASYNCcnt) overlaps the 8 WMMAs of tile k.
//   LDS row stride 72 halves (144B): 16B aligned, bank-conflict free.
// ======================================================================
#define BT_M 128
#define BT_N 64
#define BT_K 64
#define SP   72

__device__ __forceinline__ v16h frag_a(const _Float16* sA, int row, int kc, int g) {
  // ISA 7.12.2 A 16x32 layout: lane = M row; halves j<8 -> K=kc+g*8+j,
  // j>=8 -> K = kc+16+g*8+(j-8)
  v16h a;
  #pragma unroll
  for (int j = 0; j < 8; ++j) {
    a[j]     = sA[row * SP + kc + g * 8 + j];
    a[j + 8] = sA[row * SP + kc + 16 + g * 8 + j];
  }
  return a;
}

__device__ __forceinline__ v16h frag_b(const _Float16* sB, int col, int kc, int g) {
  // B 32x16 layout: lane = N col; halves j<8 -> K=kc+g*8+j,
  // j>=8 -> K = kc+16+g*8+(j-8)
  v16h b;
  #pragma unroll
  for (int j = 0; j < 8; ++j) {
    b[j]     = sB[col * SP + kc + g * 8 + j];
    b[j + 8] = sB[col * SP + kc + 16 + g * 8 + j];
  }
  return b;
}

__device__ __forceinline__ void compute_tile(const _Float16* cA, const _Float16* cB,
                                             int m0, int n0, int lr, int g,
                                             v8f& acc00, v8f& acc01,
                                             v8f& acc10, v8f& acc11) {
  #pragma unroll
  for (int kc = 0; kc < BT_K; kc += 32) {
    const v16h a0 = frag_a(cA, m0 + lr,      kc, g);
    const v16h a1 = frag_a(cA, m0 + 16 + lr, kc, g);
    const v16h b0 = frag_b(cB, n0 + lr,      kc, g);
    const v16h b1 = frag_b(cB, n0 + 16 + lr, kc, g);
    acc00 = __builtin_amdgcn_wmma_f32_16x16x32_f16(false, a0, false, b0,
                                                   (short)0, acc00, false, false);
    acc01 = __builtin_amdgcn_wmma_f32_16x16x32_f16(false, a0, false, b1,
                                                   (short)0, acc01, false, false);
    acc10 = __builtin_amdgcn_wmma_f32_16x16x32_f16(false, a1, false, b0,
                                                   (short)0, acc10, false, false);
    acc11 = __builtin_amdgcn_wmma_f32_16x16x32_f16(false, a1, false, b1,
                                                   (short)0, acc11, false, false);
  }
}

__device__ __forceinline__ void store_tile(const v8f& acc, float* __restrict__ C,
                                           const float* __restrict__ addend,
                                           int mbase, int n, int N, int g) {
  // C/D layout: VGPR r holds M = mbase + g*8 + r, N = n (lane&15 folded in)
  if (n < N) {
    #pragma unroll
    for (int r = 0; r < 8; ++r) {
      const size_t idx = (size_t)(mbase + g * 8 + r) * N + n;
      float v = acc[r];
      if (addend) v += addend[idx];
      C[idx] = v;
    }
  }
}

#if HAVE_ASYNC_LDS
// Stage one 128x64 A tile + 64x64 B tile; every thread issues EXACTLY 6
// async b128 ops (B row clamped to N-1: WMMA column n depends only on B
// column n, and columns >= N are discarded at store time), so a per-wave
// s_wait_asynccnt 6 after issuing the next tile drains exactly this tile.
__device__ __forceinline__ void stage_tile_async(
    const _Float16* __restrict__ A, const _Float16* __restrict__ W,
    _Float16* dA, _Float16* dB,
    int blockM, int blockN, int kt, int N, int K, int t) {
  #pragma unroll
  for (int i = 0; i < 4; ++i) {
    const int f  = t + i * 256;
    const int ar = f >> 3;
    const int ac = (f & 7) << 3;
    __builtin_amdgcn_global_load_async_to_lds_b128(
        (gptr_b128)&A[(size_t)(blockM + ar) * K + kt + ac],
        (lptr_b128)&dA[ar * SP + ac], 0, 0);
  }
  #pragma unroll
  for (int i = 0; i < 2; ++i) {
    const int f  = t + i * 256;
    const int br = f >> 3;
    const int bc = (f & 7) << 3;
    int row = blockN + br;
    if (row >= N) row = N - 1;   // clamp: keeps address valid, count uniform
    __builtin_amdgcn_global_load_async_to_lds_b128(
        (gptr_b128)&W[(size_t)row * K + kt + bc],
        (lptr_b128)&dB[br * SP + bc], 0, 0);
  }
}

__device__ __forceinline__ void wait_async_tile(bool more_in_flight) {
#if __has_builtin(__builtin_amdgcn_s_wait_asynccnt)
  if (more_in_flight) __builtin_amdgcn_s_wait_asynccnt(6);
  else                __builtin_amdgcn_s_wait_asynccnt(0);
#else
  if (more_in_flight) asm volatile("s_wait_asynccnt 0x6" ::: "memory");
  else                asm volatile("s_wait_asynccnt 0x0" ::: "memory");
#endif
}
#endif

template <int K>
__global__ __launch_bounds__(256)
void k_wmma_gemm(const _Float16* __restrict__ A, const _Float16* __restrict__ W,
                 float* __restrict__ C, const float* __restrict__ addend,
                 int M, int N) {
  __shared__ __align__(16) _Float16 sA[2][BT_M * SP];   // double buffered
  __shared__ __align__(16) _Float16 sB[2][BT_N * SP];

  const int t    = threadIdx.x;
  const int lane = t & 31;
  const int wave = t >> 5;
  const int m0   = (wave & 3) * 32;   // wave tile origin in block tile
  const int n0   = (wave >> 2) * 32;
  const int blockM = blockIdx.x * BT_M;
  const int blockN = blockIdx.y * BT_N;

  const int g  = lane >> 4;   // lane group
  const int lr = lane & 15;   // row (A) / col (B) within 16

  v8f acc00 = {}, acc01 = {}, acc10 = {}, acc11 = {};

  constexpr int nT = K / BT_K;

#if HAVE_ASYNC_LDS
  // ---------------- async double-buffered pipeline ----------------
  stage_tile_async(A, W, sA[0], sB[0], blockM, blockN, 0, N, K, t);
  #pragma unroll 2
  for (int it = 0; it < nT; ++it) {
    __syncthreads();   // everyone done computing tile it-1 (buffer (it+1)&1)
    if (it + 1 < nT) {
      stage_tile_async(A, W, sA[(it + 1) & 1], sB[(it + 1) & 1],
                       blockM, blockN, (it + 1) * BT_K, N, K, t);
      // L2 prefetch two tiles ahead (global_prefetch_b8)
      if (it + 2 < nT)
        __builtin_prefetch(&A[(size_t)(blockM + (t >> 1)) * K + (it + 2) * BT_K],
                           0, 1);
    }
    wait_async_tile(it + 1 < nT);   // tile it resident; tile it+1 in flight
    __syncthreads();                // all waves' portions of tile it landed
    compute_tile(sA[it & 1], sB[it & 1], m0, n0, lr, g,
                 acc00, acc01, acc10, acc11);
  }
#else
  // ---------------- fallback: register-staged single buffer ----------------
  #pragma unroll 2
  for (int it = 0; it < nT; ++it) {
    const int kt = it * BT_K;
    float4 ta[4], tb[2];
    #pragma unroll
    for (int i = 0; i < 4; ++i) {
      const int f  = t + i * 256;
      const int ar = f >> 3;
      const int ac = (f & 7) << 3;
      ta[i] = *(const float4*)&A[(size_t)(blockM + ar) * K + kt + ac];
    }
    #pragma unroll
    for (int i = 0; i < 2; ++i) {
      const int f  = t + i * 256;
      const int br = f >> 3;
      const int bc = (f & 7) << 3;
      int row = blockN + br;
      if (row >= N) row = N - 1;
      tb[i] = *(const float4*)&W[(size_t)row * K + kt + bc];
    }
    __syncthreads();   // previous tile fully consumed
    #pragma unroll
    for (int i = 0; i < 4; ++i) {
      const int f = t + i * 256;
      *(float4*)&sA[0][(f >> 3) * SP + ((f & 7) << 3)] = ta[i];
    }
    #pragma unroll
    for (int i = 0; i < 2; ++i) {
      const int f = t + i * 256;
      *(float4*)&sB[0][(f >> 3) * SP + ((f & 7) << 3)] = tb[i];
    }
    __syncthreads();
    if (it + 1 < nT)
      __builtin_prefetch(&A[(size_t)(blockM + (t >> 1)) * K + kt + BT_K], 0, 1);
    compute_tile(sA[0], sB[0], m0, n0, lr, g, acc00, acc01, acc10, acc11);
  }
#endif

  const int nA = blockN + n0 + lr;
  const int nB = blockN + n0 + 16 + lr;
  store_tile(acc00, C, addend, blockM + m0,      nA, N, g);
  store_tile(acc01, C, addend, blockM + m0,      nB, N, g);
  store_tile(acc10, C, addend, blockM + m0 + 16, nA, N, g);
  store_tile(acc11, C, addend, blockM + m0 + 16, nB, N, g);
}

// ======================================================================
// causal depthwise conv1d (k=4) + bias + SiLU over u = xz[:, :512]
// ======================================================================
__global__ void k_conv_silu(const float* __restrict__ xz,
                            const float* __restrict__ cw,
                            const float* __restrict__ cb,
                            float* __restrict__ ucf,
                            _Float16* __restrict__ uc16) {
  const int idx = blockIdx.x * blockDim.x + threadIdx.x;
  if (idx >= M_TOTAL * D_INNER) return;
  const int d  = idx & (D_INNER - 1);
  const int bl = idx / D_INNER;          // b*L + l
  const int l  = bl & (L_SEQ - 1);
  const int base = bl - l;               // b*L
  float acc = cb[d];
  #pragma unroll
  for (int k = 0; k < D_CONV; ++k) {
    const int ls = l - (D_CONV - 1) + k;
    if (ls >= 0)
      acc += cw[d * D_CONV + k] * xz[(size_t)(base + ls) * (2 * D_INNER) + d];
  }
  const float sv = acc / (1.f + __expf(-acc));   // SiLU
  ucf[idx]  = sv;
  uc16[idx] = (_Float16)sv;
}

// ======================================================================
// dt = softplus(dbc[:, :16] @ dtproj_w^T + dtproj_b); thread per (m,d)
// ======================================================================
__global__ void k_dtproj(const float* __restrict__ dbc,
                         const float* __restrict__ dw,
                         const float* __restrict__ db,
                         float* __restrict__ dt) {
  const int idx = blockIdx.x * blockDim.x + threadIdx.x;
  if (idx >= M_TOTAL * D_INNER) return;
  const int d = idx & (D_INNER - 1);
  const int m = idx >> 9;
  float acc = db[d];
  const float* r = dbc + (size_t)m * (DT_RANK + 2 * D_STATE);
  const float* w = dw + (size_t)d * DT_RANK;
  #pragma unroll
  for (int k = 0; k < DT_RANK; ++k) acc += r[k] * w[k];
  dt[idx] = (acc > 20.f) ? acc : log1pf(__expf(acc));   // softplus
}

// ======================================================================
// selective scan: lane per (b, d, s); recurrent state in ONE register;
// 16-lane shfl_xor tree reduces y = sum_s h*C_s; epilogue fuses
// y = (y + u*Dp) * silu(z) and stores f16 for the out_proj WMMA GEMM.
// ======================================================================
__global__ __launch_bounds__(256)
void k_scan(const float* __restrict__ dt, const float* __restrict__ ucf,
            const float* __restrict__ dbc, const float* __restrict__ xz,
            const float* __restrict__ A_log, const float* __restrict__ Dp,
            _Float16* __restrict__ yg16) {
  const int gtid = blockIdx.x * 256 + threadIdx.x;
  const int s  = gtid & 15;
  const int ch = gtid >> 4;             // 0 .. B*D_INNER-1
  const int d  = ch & (D_INNER - 1);
  const int b  = ch >> 9;
  const float As = -__expf(A_log[d * D_STATE + s]);   // A = -exp(A_log)
  const float Dv = Dp[d];
  float h = 0.f;
  const size_t rbase = (size_t)b * L_SEQ;
  for (int l = 0; l < L_SEQ; ++l) {
    const size_t m = rbase + l;
    const float dtv = dt[m * D_INNER + d];
    const float uv  = ucf[m * D_INNER + d];
    const float Bv  = dbc[m * (DT_RANK + 2 * D_STATE) + DT_RANK + s];
    const float Cv  = dbc[m * (DT_RANK + 2 * D_STATE) + DT_RANK + D_STATE + s];
    const float dA = __expf(dtv * As);
    h = h * dA + dtv * uv * Bv;
    float part = h * Cv;
    part += __shfl_xor(part, 1, 16);
    part += __shfl_xor(part, 2, 16);
    part += __shfl_xor(part, 4, 16);
    part += __shfl_xor(part, 8, 16);
    if (s == 0) {
      float y = part + uv * Dv;
      const float zv = xz[m * (2 * D_INNER) + D_INNER + d];
      y *= zv / (1.f + __expf(-zv));                  // * silu(z)
      yg16[m * D_INNER + d] = (_Float16)y;
    }
  }
}

// ======================================================================
// host-side orchestration (graph-capture safe: stream-only, no alloc/sync)
// ======================================================================
extern "C" void kernel_launch(void* const* d_in, const int* in_sizes, int n_in,
                              void* d_out, int out_size, void* d_ws, size_t ws_size,
                              hipStream_t stream) {
  const float* x        = (const float*)d_in[0];
  const float* ln_g     = (const float*)d_in[1];
  const float* ln_b     = (const float*)d_in[2];
  const float* in_w     = (const float*)d_in[3];
  const float* conv_w   = (const float*)d_in[4];
  const float* conv_b   = (const float*)d_in[5];
  const float* xproj_w  = (const float*)d_in[6];
  const float* dtproj_w = (const float*)d_in[7];
  const float* dtproj_b = (const float*)d_in[8];
  const float* A_log    = (const float*)d_in[9];
  const float* Dp       = (const float*)d_in[10];
  const float* out_w    = (const float*)d_in[11];
  float* out = (float*)d_out;

  // ---- workspace layout (all buffers fully rewritten every call) ----
  size_t off = 0;
  auto alloc = [&](size_t bytes) -> void* {
    void* p = (char*)d_ws + off;
    off += (bytes + 255) & ~(size_t)255;
    return p;
  };
  float*     h    = (float*)alloc((size_t)M_TOTAL * D_MODEL * 4);       // residual
  float*     xz   = (float*)alloc((size_t)M_TOTAL * 2 * D_INNER * 4);   // in_proj out
  float*     ucf  = (float*)alloc((size_t)M_TOTAL * D_INNER * 4);       // conv+silu f32
  float*     dbcb = (float*)alloc((size_t)M_TOTAL * 48 * 4);            // dt|B|C
  float*     dtb  = (float*)alloc((size_t)M_TOTAL * D_INNER * 4);       // softplus dt
  _Float16*  hn16 = (_Float16*)alloc((size_t)M_TOTAL * D_MODEL * 2);    // LN out f16
  _Float16*  uc16 = (_Float16*)alloc((size_t)M_TOTAL * D_INNER * 2);    // conv out f16
  _Float16*  yg16 = (_Float16*)alloc((size_t)M_TOTAL * D_INNER * 2);    // gated y f16
  _Float16*  w16  = (_Float16*)alloc((size_t)1024 * 256 * 2);           // weight f16

  // residual stream starts as x (inputs must not be mutated)
  (void)hipMemcpyAsync(h, x, (size_t)M_TOTAL * D_MODEL * 4,
                       hipMemcpyDeviceToDevice, stream);

  const int EW = M_TOTAL * D_INNER;   // elementwise work size

  for (int layer = 0; layer < N_LAYERS; ++layer) {
    // 1) LayerNorm -> f16
    k_layernorm<<<M_TOTAL, 256, 0, stream>>>(h, ln_g, ln_b, hn16);

    // 2) in_proj: [16384,256] x [1024,256]^T -> xz [16384,1024]
    {
      const int n = 2 * D_INNER * D_MODEL;
      k_f32_to_f16<<<(n + 255) / 256, 256, 0, stream>>>(
          in_w + (size_t)layer * n, w16, n);
      k_wmma_gemm<D_MODEL><<<dim3(M_TOTAL / BT_M, (2 * D_INNER) / BT_N),
                             256, 0, stream>>>(
          hn16, w16, xz, nullptr, M_TOTAL, 2 * D_INNER);
    }

    // 3) causal depthwise conv + SiLU
    k_conv_silu<<<(EW + 255) / 256, 256, 0, stream>>>(
        xz, conv_w + (size_t)layer * D_INNER * D_CONV,
        conv_b + (size_t)layer * D_INNER, ucf, uc16);

    // 4) xproj: [16384,512] x [48,512]^T -> dbc [16384,48]
    {
      const int n = (DT_RANK + 2 * D_STATE) * D_INNER;
      k_f32_to_f16<<<(n + 255) / 256, 256, 0, stream>>>(
          xproj_w + (size_t)layer * n, w16, n);
      k_wmma_gemm<D_INNER><<<dim3(M_TOTAL / BT_M, 1), 256, 0, stream>>>(
          uc16, w16, dbcb, nullptr, M_TOTAL, DT_RANK + 2 * D_STATE);
    }

    // 5) dt projection + softplus
    k_dtproj<<<(EW + 255) / 256, 256, 0, stream>>>(
        dbcb, dtproj_w + (size_t)layer * D_INNER * DT_RANK,
        dtproj_b + (size_t)layer * D_INNER, dtb);

    // 6) selective scan + D skip + SiLU(z) gate -> yg16
    k_scan<<<(B_SZ * D_INNER * D_STATE) / 256, 256, 0, stream>>>(
        dtb, ucf, dbcb, xz,
        A_log + (size_t)layer * D_INNER * D_STATE,
        Dp + (size_t)layer * D_INNER, yg16);

    // 7) out_proj + residual: [16384,512] x [256,512]^T + h
    {
      const int n = D_MODEL * D_INNER;
      k_f32_to_f16<<<(n + 255) / 256, 256, 0, stream>>>(
          out_w + (size_t)layer * n, w16, n);
      float* dst = (layer == N_LAYERS - 1) ? out : h;
      k_wmma_gemm<D_INNER><<<dim3(M_TOTAL / BT_M, D_MODEL / BT_N),
                             256, 0, stream>>>(
          yg16, w16, dst, h, M_TOTAL, D_MODEL);
    }
  }
}
